// RRN_22694607192274
// MI455X (gfx1250) — matrix operand order repaired and checked by reading, not compiled, not run on hardware
//
#include <hip/hip_runtime.h>

typedef _Float16 v16h __attribute__((ext_vector_type(16)));
typedef float    v8f  __attribute__((ext_vector_type(8)));
typedef unsigned v2u  __attribute__((ext_vector_type(2)));

#define N_CELLS   64
#define HID       16
#define N_EDGES   1152
#define N_STEPS   16
#define TPB       256
#define MLP_TILES 45                    // 6 (L1) + 18 (L2) + 18 (L3) + 3 (L4)
#define TOT_TILES (3 * MLP_TILES + 4)   // inp, msg, comb MLPs + 4 LSTM gate tiles = 139
#define TILE_HALFS 512                  // 32 lanes * 16 halves
#define BIAS_FLOATS (304 * 3 + 64)      // 3 MLP bias packs + fused lstm bias

// Stop the machine scheduler from hoisting weight-tile loads across layer
// boundaries / loop iterations (keeps the wave under 256 VGPRs).
#define SCHED_FENCE() __builtin_amdgcn_sched_barrier(0)

// ---------------------------------------------------------------------------
// WMMA helpers.  D = A(16x32, f16) x B(32x16, f16) + C(f32).
// A layout (as packed by rrn_prepack, contiguous K): M = lane&15, half h holds
//   K = h for lanes 0-15, K = 16+h for lanes 16-31 (same order as B).
// B layout (contiguous K, matching A): lanes 0-15 hold K=0..15 (half h = K),
//   lanes 16-31 hold K=16..31, N = lane&15.
// D layout: acc reg r, lane: M = r + 8*(lane>=16), N = lane&15.
// ---------------------------------------------------------------------------
__device__ __forceinline__ v8f wmma16(v16h a, v16h b, v8f c) {
  return __builtin_amdgcn_wmma_f32_16x16x32_f16(false, a, false, b,
                                                (short)0, c, false, false);
}

__device__ __forceinline__ v16h ldtile(const _Float16* __restrict__ t,
                                       int idx, int lane) {
  return *(const v16h*)(t + (size_t)idx * TILE_HALFS + lane * 16);
}

// v_permlane16_swap_b32: exchanges lanes 16-31 of the first operand with
// lanes 0-15 of the second.  Result: sw[0] = {a[0:15], b[0:15]},
// sw[1] = {a[16:31], b[16:31]} -- both B-tile halves in one instruction.
__device__ __forceinline__ v2u swap16pair(float a, float b) {
  return __builtin_amdgcn_permlane16_swap(__float_as_uint(a),
                                          __float_as_uint(b), false, false);
}

// Fuse two D tiles (features 0..15 / 16..31) into one B tile (K=0..31):
// b[r]   needs lane<16: a0 own-lane, lane>=16: a1 from lane-16
// b[r+8] needs lane<16: a0 from lane+16, lane>=16: a1 own-lane
// == exactly one v_permlane16_swap_b32 per register pair.
__device__ __forceinline__ v16h pack_pair(v8f a0, v8f a1) {
  v16h b;
#pragma unroll
  for (int r = 0; r < 8; ++r) {
    v2u sw = swap16pair(a0[r], a1[r]);
    b[r]     = (_Float16)__uint_as_float(sw[0]);
    b[r + 8] = (_Float16)__uint_as_float(sw[1]);
  }
  return b;
}

// 4-layer MLP  [32 -> 96 -> 96 -> 96 -> 16], weights pre-packed as A tiles.
// Pairwise structure + sched fences: at most 2 f32 accumulators, one A tile
// and 6 B tiles live at any point.
__device__ __forceinline__ v8f run_mlp(const _Float16* __restrict__ tiles,
                                       const float* __restrict__ bias,
                                       v16h b0, int lane) {
  const bool lo  = lane < 16;
  const int  hi8 = lo ? 0 : 8;
  const v8f z = {0.f, 0.f, 0.f, 0.f, 0.f, 0.f, 0.f, 0.f};
  v16h Ba[3], Bb[3];

  // L1: K=32 (input), produce Ba
#pragma unroll
  for (int p = 0; p < 3; ++p) {
    v8f a0 = wmma16(ldtile(tiles, 2 * p,     lane), b0, z);
    v8f a1 = wmma16(ldtile(tiles, 2 * p + 1, lane), b0, z);
#pragma unroll
    for (int r = 0; r < 8; ++r) {
      a0[r] = fmaxf(a0[r] + bias[p * 32 +      hi8 + r], 0.f);
      a1[r] = fmaxf(a1[r] + bias[p * 32 + 16 + hi8 + r], 0.f);
    }
    Ba[p] = pack_pair(a0, a1);
    SCHED_FENCE();
  }

  // L2: Ba -> Bb, tiles 6..23, bias offset 96
#pragma unroll
  for (int p = 0; p < 3; ++p) {
    v8f a0 = z, a1 = z;
#pragma unroll
    for (int kt = 0; kt < 3; ++kt)
      a0 = wmma16(ldtile(tiles, 6 + (2 * p) * 3 + kt, lane), Ba[kt], a0);
#pragma unroll
    for (int kt = 0; kt < 3; ++kt)
      a1 = wmma16(ldtile(tiles, 6 + (2 * p + 1) * 3 + kt, lane), Ba[kt], a1);
#pragma unroll
    for (int r = 0; r < 8; ++r) {
      a0[r] = fmaxf(a0[r] + bias[96 + p * 32 +      hi8 + r], 0.f);
      a1[r] = fmaxf(a1[r] + bias[96 + p * 32 + 16 + hi8 + r], 0.f);
    }
    Bb[p] = pack_pair(a0, a1);
    SCHED_FENCE();
  }

  // L3: Bb -> Ba, tiles 24..41, bias offset 192
#pragma unroll
  for (int p = 0; p < 3; ++p) {
    v8f a0 = z, a1 = z;
#pragma unroll
    for (int kt = 0; kt < 3; ++kt)
      a0 = wmma16(ldtile(tiles, 24 + (2 * p) * 3 + kt, lane), Bb[kt], a0);
#pragma unroll
    for (int kt = 0; kt < 3; ++kt)
      a1 = wmma16(ldtile(tiles, 24 + (2 * p + 1) * 3 + kt, lane), Bb[kt], a1);
#pragma unroll
    for (int r = 0; r < 8; ++r) {
      a0[r] = fmaxf(a0[r] + bias[192 + p * 32 +      hi8 + r], 0.f);
      a1[r] = fmaxf(a1[r] + bias[192 + p * 32 + 16 + hi8 + r], 0.f);
    }
    Ba[p] = pack_pair(a0, a1);
    SCHED_FENCE();
  }

  // L4: tiles 42..44, bias offset 288, no relu
  v8f c = z;
#pragma unroll
  for (int kt = 0; kt < 3; ++kt)
    c = wmma16(ldtile(tiles, 42 + kt, lane), Ba[kt], c);
#pragma unroll
  for (int r = 0; r < 8; ++r) c[r] += bias[288 + hi8 + r];
  SCHED_FENCE();
  return c;
}

__device__ __forceinline__ float sigf(float x) { return 1.f / (1.f + __expf(-x)); }

// ---------------------------------------------------------------------------
// Weight pre-pack: fp32 row-major (out,in) -> f16 WMMA-A register layout.
// One block per tile; last block packs biases (lstm b_ih+b_hh fused).
// ---------------------------------------------------------------------------
struct PK {
  const float* w[3][4];   // [inp,msg,comb][layer] weight matrices
  const float* b[3][4];   // biases
  const float* wih; const float* whh;
  const float* bih; const float* bhh;
};

__global__ __launch_bounds__(TPB) void rrn_prepack(PK p,
                                                   _Float16* __restrict__ pre,
                                                   float* __restrict__ wsb) {
  const int g = blockIdx.x;
  if (g < TOT_TILES) {
    for (int idx = threadIdx.x; idx < TILE_HALFS; idx += blockDim.x) {
      const int lane = idx >> 4;
      const int h = idx & 15;
      const int j = h >> 1, hk = h & 1;
      const int M = lane & 15;
      const int K = ((lane >= 16) ? 16 : 0) + h; (void)j; (void)hk;
      float val = 0.f;
      if (g < 3 * MLP_TILES) {
        const int grp = g / MLP_TILES;
        const int t = g % MLP_TILES;
        int layer, nt, kt;
        if (t < 6)       { layer = 0; nt = t;            kt = 0; }
        else if (t < 24) { layer = 1; nt = (t - 6) / 3;  kt = (t - 6) % 3; }
        else if (t < 42) { layer = 2; nt = (t - 24) / 3; kt = (t - 24) % 3; }
        else             { layer = 3; nt = 0;            kt = t - 42; }
        const int Kin = (layer == 0) ? ((grp == 0) ? 25 : 32) : 96;
        const float* W = p.w[grp][layer];
        const int Mg = nt * 16 + M;
        const int Kg = kt * 32 + K;
        if (Kg < Kin) val = W[Mg * Kin + Kg];
      } else {
        // LSTM gate tile: A(16 gates x 32) = [W_ih | W_hh]
        const int t = g - 3 * MLP_TILES;
        const int Mg = t * 16 + M;
        val = (K < 16) ? p.wih[Mg * 16 + K] : p.whh[Mg * 16 + (K - 16)];
      }
      pre[(size_t)g * TILE_HALFS + idx] = (_Float16)val;
    }
  } else {
    for (int i = threadIdx.x; i < 912; i += blockDim.x) {
      const int grp = i / 304;
      const int o = i % 304;
      int layer, off;
      if (o < 96)       { layer = 0; off = 0;   }
      else if (o < 192) { layer = 1; off = 96;  }
      else if (o < 288) { layer = 2; off = 192; }
      else              { layer = 3; off = 288; }
      wsb[i] = p.b[grp][layer][o - off];
    }
    for (int i = threadIdx.x; i < 64; i += blockDim.x)
      wsb[912 + i] = p.bih[i] + p.bhh[i];
  }
}

// ---------------------------------------------------------------------------
// Main fused kernel: one block (8 waves, wave32) per batch element.
// H (f16), C (f32), X0 (f16), agg (f32) and edge lists live in LDS for all
// 16 recurrent steps; msg scatter-sum via ds_add_f32 atomics.
// ---------------------------------------------------------------------------
__global__ __launch_bounds__(TPB)
__attribute__((amdgpu_num_vgpr(224)))
void rrn_main(
    const float* __restrict__ X, const _Float16* __restrict__ pre,
    const float* __restrict__ wsb, const int* __restrict__ lg,
    const int* __restrict__ rg, const float* __restrict__ Wd,
    const float* __restrict__ bd, float* __restrict__ out) {
  __shared__ __align__(32) _Float16 Hs[N_CELLS * HID];
  __shared__ __align__(32) _Float16 X0s[N_CELLS * HID];
  __shared__ float Cs[N_CELLS * HID];
  __shared__ float aggs[N_CELLS * HID];
  __shared__ short ls[N_EDGES], rs[N_EDGES];
  __shared__ float bsh[BIAS_FLOATS];

  const int tid = threadIdx.x;
  const int lane = tid & 31;
  const int wave = tid >> 5;
  const bool lo = lane < 16;
  const int hi8 = lo ? 0 : 8;
  const int n = lane & 15;
  const int bi = blockIdx.x;

  const _Float16* pre_inp  = pre;
  const _Float16* pre_msg  = pre + 45 * TILE_HALFS;
  const _Float16* pre_comb = pre + 90 * TILE_HALFS;
  const _Float16* pre_lstm = pre + 135 * TILE_HALFS;
  const float* b_inp  = bsh;
  const float* b_msg  = bsh + 304;
  const float* b_comb = bsh + 608;
  const float* b_lstm = bsh + 912;

  for (int i = tid; i < N_EDGES; i += TPB) {
    ls[i] = (short)lg[i];
    rs[i] = (short)rg[i];
  }
  for (int i = tid; i < BIAS_FLOATS; i += TPB) bsh[i] = wsb[i];
  for (int i = tid; i < N_CELLS * HID; i += TPB) Cs[i] = 0.f;
  __syncthreads();

  // ---- input encoder: X0 = MLP(concat(RC, X)) ; all waves compute (dup x2),
  //      stores predicated so WMMA always runs with full EXEC.
  {
    const int ct = wave & 3;
    const int cell = ct * 16 + n;
    v16h b0;
    if (lo) {
      // K<8: onehot(cell%8), K 8..15: onehot(cell/8)
#pragma unroll
      for (int h = 0; h < 8; ++h)
        b0[h] = ((cell & 7) == h) ? (_Float16)1.f : (_Float16)0.f;
#pragma unroll
      for (int h = 8; h < 16; ++h)
        b0[h] = ((cell >> 3) == (h - 8)) ? (_Float16)1.f : (_Float16)0.f;
    } else {
      const float* xp = X + ((size_t)bi * N_CELLS + cell) * 9;
#pragma unroll
      for (int h = 0; h < 16; ++h)
        b0[h] = (h < 9) ? (_Float16)xp[h] : (_Float16)0.f;
    }
    v8f x0 = run_mlp(pre_inp, b_inp, b0, lane);
    if (wave < 4) {
#pragma unroll
      for (int r = 0; r < 8; ++r) {
        _Float16 v = (_Float16)x0[r];
        Hs[cell * HID + hi8 + r] = v;
        X0s[cell * HID + hi8 + r] = v;
      }
    }
  }
  __syncthreads();

  // ---- 16 recurrent steps
  for (int step = 0; step < N_STEPS; ++step) {
    for (int i = tid; i < N_CELLS * HID; i += TPB) aggs[i] = 0.f;
    __syncthreads();

    // message MLP over 72 edge tiles (9 per wave), scatter-sum into aggs
    for (int et = wave; et < 72; et += 8) {
      SCHED_FENCE();  // no pipelining across edge tiles
      const int e = et * 16 + n;
      const int cell = lo ? (int)ls[e] : (int)rs[e];
      v16h b0 = *(const v16h*)&Hs[cell * HID];  // K<16: H[l], K>=16: H[r]
      v8f m = run_mlp(pre_msg, b_msg, b0, lane);
      const int recv = (int)rs[e];
#pragma unroll
      for (int r = 0; r < 8; ++r)
        atomicAdd(&aggs[recv * HID + hi8 + r], m[r]);
    }
    __syncthreads();

    // combine MLP + LSTM cell over 4 cell tiles (all waves compute, dup x2)
    {
      const int ct = wave & 3;
      const int cell = ct * 16 + n;
      v16h xb = *(const v16h*)&X0s[cell * HID];
      v16h b0;
#pragma unroll
      for (int h = 0; h < 16; ++h)
        b0[h] = lo ? xb[h] : (_Float16)aggs[cell * HID + h];
      v8f xm = run_mlp(pre_comb, b_comb, b0, lane);

      // LSTM input B tile: K<16 = XM features, K>=16 = H features
      v16h hb = *(const v16h*)&Hs[cell * HID];
      v16h bl;
#pragma unroll
      for (int r = 0; r < 8; ++r) {
        v2u sw = swap16pair(xm[r], xm[r]);
        bl[r]     = lo ? (_Float16)__uint_as_float(sw[0]) : hb[r];
        bl[r + 8] = lo ? (_Float16)__uint_as_float(sw[1]) : hb[r + 8];
      }
      const v8f z = {0.f, 0.f, 0.f, 0.f, 0.f, 0.f, 0.f, 0.f};
      v8f gi = wmma16(ldtile(pre_lstm, 0, lane), bl, z);
      v8f gf = wmma16(ldtile(pre_lstm, 1, lane), bl, z);
      v8f gg = wmma16(ldtile(pre_lstm, 2, lane), bl, z);
      v8f go = wmma16(ldtile(pre_lstm, 3, lane), bl, z);
      if (wave < 4) {
#pragma unroll
        for (int r = 0; r < 8; ++r) {
          const int u = hi8 + r;
          const int idx = cell * HID + u;
          float iv = sigf(gi[r] + b_lstm[0 + u]);
          float fv = sigf(gf[r] + b_lstm[16 + u]);
          float gv = tanhf(gg[r] + b_lstm[32 + u]);
          float ov = sigf(go[r] + b_lstm[48 + u]);
          float c = fv * Cs[idx] + iv * gv;
          Cs[idx] = c;
          Hs[idx] = (_Float16)(ov * tanhf(c));
        }
      }
    }
    __syncthreads();
  }

  // ---- decoder: out[bi*64+cell, j] = H[cell] . Wd[j] + bd[j]
  for (int idx = tid; idx < N_CELLS * 8; idx += TPB) {
    const int cell = idx >> 3, j = idx & 7;
    float s = bd[j];
#pragma unroll
    for (int u = 0; u < HID; ++u)
      s += (float)Hs[cell * HID + u] * Wd[j * HID + u];
    out[((size_t)bi * N_CELLS + cell) * 8 + j] = s;
  }
}

// ---------------------------------------------------------------------------
// d_in leaf order: X, inp_enc(W1,b1..W4,b4), msg_enc(8), msg_comb(8),
// lstm(W_ih,W_hh,b_ih,b_hh), decoder(Wd,bd), l, r
// ---------------------------------------------------------------------------
extern "C" void kernel_launch(void* const* d_in, const int* in_sizes, int n_in,
                              void* d_out, int out_size, void* d_ws,
                              size_t ws_size, hipStream_t stream) {
  (void)in_sizes; (void)n_in; (void)out_size; (void)ws_size;

  const float* X = (const float*)d_in[0];
  PK p;
  for (int m = 0; m < 3; ++m)
    for (int l = 0; l < 4; ++l) {
      p.w[m][l] = (const float*)d_in[1 + m * 8 + l * 2];
      p.b[m][l] = (const float*)d_in[1 + m * 8 + l * 2 + 1];
    }
  p.wih = (const float*)d_in[25];
  p.whh = (const float*)d_in[26];
  p.bih = (const float*)d_in[27];
  p.bhh = (const float*)d_in[28];
  const float* Wd = (const float*)d_in[29];
  const float* bd = (const float*)d_in[30];
  const int* lg = (const int*)d_in[31];
  const int* rg = (const int*)d_in[32];

  _Float16* pre = (_Float16*)d_ws;
  float* wsb = (float*)((char*)d_ws + (size_t)TOT_TILES * TILE_HALFS * sizeof(_Float16));

  rrn_prepack<<<TOT_TILES + 1, TPB, 0, stream>>>(p, pre, wsb);
  rrn_main<<<64, TPB, 0, stream>>>(X, pre, wsb, lg, rg, Wd, bd, (float*)d_out);
}
